// SpikeNet_55456617726553
// MI455X (gfx1250) — matrix operand order repaired and checked by reading, not compiled, not run on hardware
//
#include <hip/hip_runtime.h>
#include <hip/hip_bf16.h>

typedef __attribute__((ext_vector_type(16))) _Float16 v16h;
typedef __attribute__((ext_vector_type(8)))  _Float16 v8h;
typedef __attribute__((ext_vector_type(8)))  float    v8f;

#define AS1 __attribute__((address_space(1)))

#define NB   256
#define NI   300
#define NIP  320   // NI padded to a multiple of 32
#define NH   512
#define NH2  64
#define NO   2
#define NT   512

static __device__ __forceinline__ v8f wmma_f16(v16h a, v16h b, v8f c) {
  // D = A(16x32 f16) * B(32x16 f16) + C(16x16 f32)
  return __builtin_amdgcn_wmma_f32_16x16x32_f16(false, a, false, b, (short)0, c, false, false);
}

// Opaque copy of a (uniform) pointer in SGPRs: blocks LICM from hoisting the
// weight-fragment loads out of the 512-step time loop (which caused the RA to
// spill every fragment to scratch). Returned as an explicit global (AS1)
// pointer so the loads select global_load_b128 (LOADcnt only) instead of
// flat_load_b128 (which would also bump DScnt and couple against the LDS
// traffic in the same loop).
static __device__ __forceinline__ const AS1 _Float16* launder(const _Float16* p) {
  asm volatile("" : "+s"(p));
  return (const AS1 _Float16*)p;
}

// A fragment (16x32 f16, row-major stride ld). Row = lane&15.
// Lane<16: K = k0+0..15 ; Lane>=16: K = k0+16..31.
// Two aligned 16-byte LDS loads (ld multiple of 16, k0 multiple of 32).
static __device__ __forceinline__ v16h load_a_h(const _Float16* src, int ld, int m, int hi, int k0) {
  const _Float16* p = src + (size_t)m * ld + k0 + (hi ? 16 : 0);
  v8h lo = *(const v8h*)p;
  v8h hb = *(const v8h*)(p + 8);
  v16h a;
#pragma unroll
  for (int h = 0; h < 8; ++h) { a[h] = lo[h]; a[h + 8] = hb[h]; }
  return a;
}

// B fragment (32x16): B[k][n] = src[n*ld + k] (weight row n, K contiguous), f16 source.
// Lane<16: K = k0+0..15 ; Lane>=16: K = k0+16..31. Column n = lane&15.
// One aligned 32-byte global load.
static __device__ __forceinline__ v16h load_b_h(const AS1 _Float16* src, int ld, int n, int hi, int k0) {
  return *(const AS1 v16h*)(src + (size_t)n * ld + k0 + (hi ? 16 : 0));
}

// ---- one-shot weight conversion kernels (f32 -> f16, L2-resident thereafter) ----
__global__ void k_cvt_pad_w1(const float* __restrict__ src, _Float16* __restrict__ dst) {
  int i = blockIdx.x * 256 + threadIdx.x;           // NH * NIP elements
  if (i >= NH * NIP) return;
  int r = i / NIP, c = i - r * NIP;
  dst[i] = (c < NI) ? (_Float16)src[r * NI + c] : (_Float16)0.f;
}
__global__ void k_cvt(const float* __restrict__ src, _Float16* __restrict__ dst, int n) {
  int i = blockIdx.x * 256 + threadIdx.x;
  if (i < n) dst[i] = (_Float16)src[i];
}

__global__ __launch_bounds__(512) void snn_persistent(
    const float* __restrict__ x,
    const _Float16* __restrict__ W1h,   // [NH  x NIP] f16, zero-padded K
    const _Float16* __restrict__ V1h,   // [NH  x NH ] f16
    const _Float16* __restrict__ W2h,   // [NH2 x NH ] f16
    const _Float16* __restrict__ V2h,   // [NH2 x NH2] f16
    const float* __restrict__ b1,  const float* __restrict__ V1b,
    const float* __restrict__ beta1, const float* __restrict__ thr1,
    const float* __restrict__ b2,  const float* __restrict__ V2b,
    const float* __restrict__ beta2, const float* __restrict__ thr2,
    const float* __restrict__ Wo,  const float* __restrict__ bo,
    const float* __restrict__ V3w, const float* __restrict__ V3b,
    const float* __restrict__ beta3,
    float* __restrict__ out)
{
  __shared__ _Float16 s_x[16 * NIP];     // 10 KB: this step's input tile (f16, padded)
  __shared__ _Float16 s_spk1[16 * NH];   // 16 KB: layer-1 spikes (persist across t)
  __shared__ _Float16 s_spk2[16 * NH2];  // 2 KB
  __shared__ float    s_spk3[16 * NO];   // 128 B
  __shared__ float    s_wo[NO * NH2];    // 512 B: output weights

  const int tid  = threadIdx.x;
  const int wave = tid >> 5;
  const int lane = tid & 31;
  const int m    = lane & 15;   // row within tile / column n for B fragments
  const int hi   = lane >> 4;
  const int b0   = blockIdx.x * 16;  // this stripe's batch rows [b0, b0+16)

  // t=0 carry state is all zeros; stage Wo
  for (int i = tid; i < 16 * NH;  i += 512) s_spk1[i] = (_Float16)0.f;
  for (int i = tid; i < 16 * NH2; i += 512) s_spk2[i] = (_Float16)0.f;
  for (int i = tid; i < 16 * NO;  i += 512) s_spk3[i] = 0.f;
  if (tid < NO * NH2) s_wo[tid] = Wo[tid];

  // Hoist per-neuron constants (column h fixed per (wave, tile, lane) for all t).
  float th1[2], cc1[2], ib1[2];
#pragma unroll
  for (int tt = 0; tt < 2; ++tt) {
    int h   = (wave + tt * 16) * 16 + m;
    th1[tt] = thr1[h];
    cc1[tt] = fminf(fmaxf(beta1[h], 0.f), 1.f);
    ib1[tt] = b1[h] + V1b[h];
  }
  float th2 = 0.f, cc2 = 0.f, ib2 = 0.f;
  if (wave < 4) {
    int h = wave * 16 + m;
    th2 = thr2[h];
    cc2 = fminf(fmaxf(beta2[h], 0.f), 1.f);
    ib2 = b2[h] + V2b[h];
  }
  const int row3 = lane >> 1, o3 = lane & 1;  // 16 rows x 2 outputs = 32 lanes of wave 0
  float ib3 = 0.f, cc3 = 0.f, v3w0 = 0.f, v3w1 = 0.f;
  if (wave == 0) {
    ib3  = bo[o3] + V3b[o3];
    cc3  = fminf(fmaxf(beta3[o3], 0.f), 1.f);
    v3w0 = V3w[o3 * 2 + 0];
    v3w1 = V3w[o3 * 2 + 1];
  }

  // Membrane state lives in VGPRs (each element owned by exactly one lane for all t).
  v8f m1[2]; v8f m2; float m3 = 0.f;
#pragma unroll
  for (int v = 0; v < 8; ++v) { m1[0][v] = 0.f; m1[1][v] = 0.f; m2[v] = 0.f; }

  __syncthreads();

  for (int t = 0; t < NT; ++t) {
    // Re-derive weight base pointers opaquely each step (global AS, SGPR base).
    const AS1 _Float16* W1t = launder(W1h);
    const AS1 _Float16* V1t = launder(V1h);
    const AS1 _Float16* W2t = launder(W2h);
    const AS1 _Float16* V2t = launder(V2h);

    // ---- stage x_t stripe into LDS (f16, zero-padded to NIP) ----
    const float* x_t = x + (size_t)t * NB * NI + (size_t)b0 * NI;
#pragma unroll
    for (int i = tid; i < 16 * NIP; i += 512) {
      int r = i / NIP, c = i - r * NIP;
      s_x[i] = (c < NI) ? (_Float16)x_t[r * NI + c] : (_Float16)0.f;
    }
    __syncthreads();                 // s_x ready

    // ---- layer 1: all 16 waves, 2 column tiles each (32 tiles = H/16) ----
    _Float16 ns1[2][8];
#pragma unroll
    for (int tt = 0; tt < 2; ++tt) {
      const int h0 = (wave + tt * 16) * 16;
      v8f acc = {};
      for (int k0 = 0; k0 < NIP; k0 += 32) {       // cur1 = x_t @ W1^T (padded, unguarded)
        v16h a  = load_a_h(s_x, NIP, m, hi, k0);
        v16h bm = load_b_h(W1t + (size_t)h0 * NIP, NIP, m, hi, k0);
        acc = wmma_f16(a, bm, acc);
      }
      for (int k0 = 0; k0 < NH; k0 += 32) {        // + spk1_prev @ V1w^T
        v16h a  = load_a_h(s_spk1, NH, m, hi, k0);
        v16h bm = load_b_h(V1t + (size_t)h0 * NH, NH, m, hi, k0);
        acc = wmma_f16(a, bm, acc);
      }
#pragma unroll
      for (int v = 0; v < 8; ++v) {
        float mo    = m1[tt][v];
        float reset = (mo > th1[tt]) ? th1[tt] : 0.f;   // subtract-reset from pre-update mem
        float mn    = cc1[tt] * mo + acc[v] + ib1[tt] - reset;
        m1[tt][v]   = mn;
        ns1[tt][v]  = (mn > th1[tt]) ? (_Float16)1.f : (_Float16)0.f;
      }
    }
    __syncthreads();                 // all waves done reading old s_spk1
#pragma unroll
    for (int tt = 0; tt < 2; ++tt) {
      const int h = (wave + tt * 16) * 16 + m;
#pragma unroll
      for (int v = 0; v < 8; ++v) s_spk1[(v + hi * 8) * NH + h] = ns1[tt][v];
    }
    __syncthreads();                 // new s_spk1 visible

    // ---- layer 2: waves 0..3 (H2/16 = 4 column tiles) ----
    _Float16 ns2[8];
    if (wave < 4) {
      const int h0 = wave * 16;
      v8f acc = {};
      for (int k0 = 0; k0 < NH; k0 += 32) {        // spk1_new @ W2^T
        v16h a  = load_a_h(s_spk1, NH, m, hi, k0);
        v16h bm = load_b_h(W2t + (size_t)h0 * NH, NH, m, hi, k0);
        acc = wmma_f16(a, bm, acc);
      }
#pragma unroll
      for (int k0 = 0; k0 < NH2; k0 += 32) {       // + spk2_prev @ V2w^T
        v16h a  = load_a_h(s_spk2, NH2, m, hi, k0);
        v16h bm = load_b_h(V2t + (size_t)h0 * NH2, NH2, m, hi, k0);
        acc = wmma_f16(a, bm, acc);
      }
#pragma unroll
      for (int v = 0; v < 8; ++v) {
        float mo    = m2[v];
        float reset = (mo > th2) ? th2 : 0.f;
        float mn    = cc2 * mo + acc[v] + ib2 - reset;
        m2[v]       = mn;
        ns2[v]      = (mn > th2) ? (_Float16)1.f : (_Float16)0.f;
      }
    }
    __syncthreads();                 // waves 0..3 done reading old s_spk2
    if (wave < 4) {
      const int h = wave * 16 + m;
#pragma unroll
      for (int v = 0; v < 8; ++v) s_spk2[(v + hi * 8) * NH2 + h] = ns2[v];
    }
    __syncthreads();                 // new s_spk2 visible

    // ---- layer 3: wave 0, O=2 -> VALU dot products (below WMMA granularity) ----
    if (wave == 0) {
      float dot = 0.f;
#pragma unroll
      for (int j = 0; j < NH2; ++j)
        dot += (float)s_spk2[row3 * NH2 + j] * s_wo[o3 * NH2 + j];
      // in-wave: both LDS reads issue before the store below (LDS in program order per wave)
      float p0 = s_spk3[row3 * NO + 0];
      float p1 = s_spk3[row3 * NO + 1];
      float mn = cc3 * m3 + dot + ib3 + p0 * v3w0 + p1 * v3w1;  // reset_mechanism='none'
      m3 = mn;
      float s = (mn > 0.1f) ? 1.f : 0.f;                         // THR3
      s_spk3[row3 * NO + o3] = s;
      out[(size_t)t * NB * NO + (size_t)(b0 + row3) * NO + o3] = s;
    }
    __syncthreads();
  }
}

extern "C" void kernel_launch(void* const* d_in, const int* in_sizes, int n_in,
                              void* d_out, int out_size, void* d_ws, size_t ws_size,
                              hipStream_t stream) {
  (void)in_sizes; (void)n_in; (void)out_size; (void)ws_size;
  const float* x     = (const float*)d_in[0];
  const float* W1    = (const float*)d_in[1];
  const float* b1    = (const float*)d_in[2];
  const float* V1w   = (const float*)d_in[3];
  const float* V1b   = (const float*)d_in[4];
  const float* beta1 = (const float*)d_in[5];
  const float* thr1  = (const float*)d_in[6];
  const float* W2    = (const float*)d_in[7];
  const float* b2    = (const float*)d_in[8];
  const float* V2w   = (const float*)d_in[9];
  const float* V2b   = (const float*)d_in[10];
  const float* beta2 = (const float*)d_in[11];
  const float* thr2  = (const float*)d_in[12];
  const float* Wo    = (const float*)d_in[13];
  const float* bo    = (const float*)d_in[14];
  const float* V3w   = (const float*)d_in[15];
  const float* V3b   = (const float*)d_in[16];
  const float* beta3 = (const float*)d_in[17];

  // Carve f16 weight buffers out of scratch (all offsets 256B-aligned; ~925 KB total).
  char* wsb = (char*)d_ws;
  _Float16* W1h = (_Float16*)(wsb);                                  // NH*NIP  = 163840 h
  _Float16* V1h = (_Float16*)(wsb + (size_t)NH * NIP * 2);           // NH*NH   = 262144 h
  _Float16* W2h = (_Float16*)(wsb + (size_t)NH * NIP * 2 + (size_t)NH * NH * 2);
  _Float16* V2h = (_Float16*)(wsb + (size_t)NH * NIP * 2 + (size_t)NH * NH * 2
                                  + (size_t)NH2 * NH * 2);

  k_cvt_pad_w1<<<dim3((NH * NIP + 255) / 256), dim3(256), 0, stream>>>(W1, W1h);
  k_cvt<<<dim3((NH * NH  + 255) / 256), dim3(256), 0, stream>>>(V1w, V1h, NH * NH);
  k_cvt<<<dim3((NH2 * NH + 255) / 256), dim3(256), 0, stream>>>(W2,  W2h, NH2 * NH);
  k_cvt<<<dim3((NH2 * NH2 + 255) / 256), dim3(256), 0, stream>>>(V2w, V2h, NH2 * NH2);

  // 16 independent batch stripes -> 16 workgroups, 16 waves each; whole T-loop in one launch.
  snn_persistent<<<dim3(16), dim3(512), 0, stream>>>(
      x, W1h, V1h, W2h, V2h,
      b1, V1b, beta1, thr1,
      b2, V2b, beta2, thr2,
      Wo, bo, V3w, V3b, beta3, (float*)d_out);
}
